// SAGE_13589276524996
// MI455X (gfx1250) — matrix operand order repaired and checked by reading, not compiled, not run on hardware
//
#include <hip/hip_runtime.h>

#define N_NODES  100000
#define N_EDGES  640000
#define IN_FEAT  128
#define HID_FEAT 128
#define OUT_FEAT 40
#define XSTR     132   // padded LDS row stride in ushorts (bank-conflict avoidance)

typedef __attribute__((ext_vector_type(16))) __bf16 v16bf;
typedef __attribute__((ext_vector_type(8)))  float  v8f;

union FragBF {
    v16bf bf;
    unsigned int u[8];
    uint4 q[2];
};

// Native bf16 conversion (hardware cvt)
__device__ __forceinline__ unsigned short f2bf(float f) {
    __bf16 b = (__bf16)f;
    return __builtin_bit_cast(unsigned short, b);
}
__device__ __forceinline__ unsigned int pack2(float a, float b) {
    return (unsigned int)f2bf(a) | ((unsigned int)f2bf(b) << 16);
}

// ---------------------------------------------------------------- zero fill
__global__ void zero_f32(float* __restrict__ p, int n) {
    int i = blockIdx.x * blockDim.x + threadIdx.x;
    if (i < n) p[i] = 0.0f;
}

// ------------------------------------- weight pre-pack into fragment order
// Wf[((kk*ntiles + t)*32 + lane)*8 + p] = bf16 pair for WMMA B-fragment:
//   k = kk*32 + 16*(lane>>4) + 2p, col = 16t + (lane&15)   (zero-pad col>=outw)
// Each lane's 8 dwords are then 32B contiguous -> two b128 loads per fragment.
__global__ void pack_weights_frag(const float* __restrict__ W,
                                  unsigned int* __restrict__ Wf,
                                  int outw, int ntiles, int total) {
    int i = blockIdx.x * blockDim.x + threadIdx.x;
    if (i >= total) return;
    int p   = i & 7;
    int L   = (i >> 3) & 31;
    int t   = (i >> 8) % ntiles;
    int kk  = (i >> 8) / ntiles;
    int k   = kk * 32 + (L >> 4) * 16 + 2 * p;
    int col = t * 16 + (L & 15);
    float a = (col < outw) ? W[k * outw + col] : 0.0f;
    float b = (col < outw) ? W[(k + 1) * outw + col] : 0.0f;
    Wf[i] = pack2(a, b);
}

// ------------------------------------------------- edge scatter (mean prep)
// One wave per edge: 32 lanes cooperatively read one contiguous 512B feature
// row (float4 each -> fully coalesced, L2-resident) and scatter-add into agg.
__global__ __launch_bounds__(256)
void scatter_edges(const float* __restrict__ feat, const long long* __restrict__ ei,
                   float* __restrict__ agg, float* __restrict__ cnt,
                   int nEdges, int addCnt) {
    int t = blockIdx.x * blockDim.x + threadIdx.x;
    int e = t >> 5;
    int lane = t & 31;
    if (e >= nEdges) return;
    long long s = ei[e];
    long long d = ei[nEdges + e];
    const float4* row = (const float4*)(feat + s * 128);
    float4 v = row[lane];
    float* o = agg + d * 128 + lane * 4;
    atomicAdd(o + 0, v.x);
    atomicAdd(o + 1, v.y);
    atomicAdd(o + 2, v.z);
    atomicAdd(o + 3, v.w);
    if (addCnt && lane == 0) atomicAdd(cnt + d, 1.0f);
}

// ------------------------------------------------------------ layer 1 GEMM
// 16 nodes per block, 8 waves; wave w computes output columns 16w..16w+15.
// h = relu( mean(agg) @ Wl + x @ Wr + b ), K = 128 (4 WMMA steps x 2 operands).
__global__ __launch_bounds__(256)
void sage_layer1(const float* __restrict__ x, const float* __restrict__ agg,
                 const float* __restrict__ cnt,
                 const uint4* __restrict__ Wlf,
                 const uint4* __restrict__ Wrf,
                 const float* __restrict__ b, float* __restrict__ h) {
    __shared__ unsigned short xs[16 * XSTR];
    __shared__ unsigned short as_[16 * XSTR];

    const int tid = threadIdx.x;
    const int node0 = blockIdx.x * 16;

    for (int i = tid; i < 16 * 128; i += 256) {
        int row = i >> 7, col = i & 127;
        int g = (node0 + row) * 128 + col;
        float inv = 1.0f / fmaxf(cnt[node0 + row], 1.0f);
        xs[row * XSTR + col]  = f2bf(x[g]);
        as_[row * XSTR + col] = f2bf(agg[g] * inv);
    }
    __syncthreads();

    const int wave  = tid >> 5;
    const int lane  = tid & 31;
    const int nloc  = lane & 15;
    const int khalf = lane >> 4;
    const int colg  = wave * 16 + nloc;

    // A-fragment per-VGPR K base offsets within a K=32 block (contiguous k = 16*khalf + s, same order as packed B)
    int kbase[8];
#pragma unroll
    for (int p = 0; p < 8; ++p)
        kbase[p] = 16 * khalf + 2 * p;

    v8f acc = {};
#pragma unroll
    for (int kk = 0; kk < 4; ++kk) {
        FragBF aA, aX, bL, bR;
#pragma unroll
        for (int p = 0; p < 8; ++p) {
            int k = kk * 32 + kbase[p];
            aA.u[p] = *(const unsigned int*)&as_[nloc * XSTR + k];
            aX.u[p] = *(const unsigned int*)&xs[nloc * XSTR + k];
        }
        // B fragment: two coalesced b128 loads from fragment-ordered weights
        int fb = ((kk * 8 + wave) * 32 + lane) * 2;   // in uint4 units
        bL.q[0] = Wlf[fb];
        bL.q[1] = Wlf[fb + 1];
        bR.q[0] = Wrf[fb];
        bR.q[1] = Wrf[fb + 1];
        acc = __builtin_amdgcn_wmma_f32_16x16x32_bf16(false, aA.bf, false, bL.bf,
                                                      (short)0, acc, false, false);
        acc = __builtin_amdgcn_wmma_f32_16x16x32_bf16(false, aX.bf, false, bR.bf,
                                                      (short)0, acc, false, false);
    }

    float bias = b[colg];
#pragma unroll
    for (int r = 0; r < 8; ++r) {
        int m = r + 8 * khalf;
        float v = fmaxf(acc[r] + bias, 0.0f);
        h[(node0 + m) * 128 + colg] = v;
    }
}

// --------------------------------------- layer 2 GEMM + fused log-softmax
// 16 nodes per block, 4 waves; waves 0..2 cover column tiles 0-15,16-31,32-47
// (cols >= 40 pre-zeroed in the packed weights). Then 16 threads row-softmax.
__global__ __launch_bounds__(128)
void sage_layer2(const float* __restrict__ h, const float* __restrict__ agg,
                 const float* __restrict__ cnt,
                 const uint4* __restrict__ Wlf,
                 const uint4* __restrict__ Wrf,
                 const float* __restrict__ b, float* __restrict__ out) {
    __shared__ unsigned short hs[16 * XSTR];
    __shared__ unsigned short as_[16 * XSTR];
    __shared__ float outs[16][48];

    const int tid = threadIdx.x;
    const int node0 = blockIdx.x * 16;

    for (int i = tid; i < 16 * 128; i += 128) {
        int row = i >> 7, col = i & 127;
        int g = (node0 + row) * 128 + col;
        float inv = 1.0f / fmaxf(cnt[node0 + row], 1.0f);
        hs[row * XSTR + col]  = f2bf(h[g]);
        as_[row * XSTR + col] = f2bf(agg[g] * inv);
    }
    __syncthreads();

    const int wave = tid >> 5;
    const int lane = tid & 31;
    if (wave < 3) {                       // wave-uniform branch: EXEC all-ones
        const int nloc  = lane & 15;
        const int khalf = lane >> 4;
        const int colg  = wave * 16 + nloc;
        const bool valid = colg < OUT_FEAT;

        int kbase[8];
#pragma unroll
        for (int p = 0; p < 8; ++p)
            kbase[p] = 16 * khalf + 2 * p;

        v8f acc = {};
#pragma unroll
        for (int kk = 0; kk < 4; ++kk) {
            FragBF aA, aH, bL, bR;
#pragma unroll
            for (int p = 0; p < 8; ++p) {
                int k = kk * 32 + kbase[p];
                aA.u[p] = *(const unsigned int*)&as_[nloc * XSTR + k];
                aH.u[p] = *(const unsigned int*)&hs[nloc * XSTR + k];
            }
            int fb = ((kk * 3 + wave) * 32 + lane) * 2;   // ntiles = 3
            bL.q[0] = Wlf[fb];
            bL.q[1] = Wlf[fb + 1];
            bR.q[0] = Wrf[fb];
            bR.q[1] = Wrf[fb + 1];
            acc = __builtin_amdgcn_wmma_f32_16x16x32_bf16(false, aA.bf, false, bL.bf,
                                                          (short)0, acc, false, false);
            acc = __builtin_amdgcn_wmma_f32_16x16x32_bf16(false, aH.bf, false, bR.bf,
                                                          (short)0, acc, false, false);
        }

        float bias = valid ? b[colg] : 0.0f;
#pragma unroll
        for (int r = 0; r < 8; ++r) {
            int m = r + 8 * khalf;
            outs[m][colg] = acc[r] + bias;
        }
    }
    __syncthreads();

    if (tid < 16) {
        int row = tid;
        float mx = -1e30f;
        for (int c = 0; c < OUT_FEAT; ++c) mx = fmaxf(mx, outs[row][c]);
        float s = 0.0f;
        for (int c = 0; c < OUT_FEAT; ++c) s += __expf(outs[row][c] - mx);
        float lse = __logf(s);
        for (int c = 0; c < OUT_FEAT; ++c)
            out[(node0 + row) * OUT_FEAT + c] = outs[row][c] - mx - lse;
    }
}

// ---------------------------------------------------------------- launcher
extern "C" void kernel_launch(void* const* d_in, const int* in_sizes, int n_in,
                              void* d_out, int out_size, void* d_ws, size_t ws_size,
                              hipStream_t stream) {
    (void)in_sizes; (void)n_in; (void)out_size; (void)ws_size;
    const float*     x   = (const float*)d_in[0];
    const long long* ei  = (const long long*)d_in[1];   // int64 (2, E)
    const float*     W1l = (const float*)d_in[2];
    const float*     W1r = (const float*)d_in[3];
    const float*     b1  = (const float*)d_in[4];
    const float*     W2l = (const float*)d_in[5];
    const float*     W2r = (const float*)d_in[6];
    const float*     b2  = (const float*)d_in[7];
    float* out = (float*)d_out;

    float* agg = (float*)d_ws;                           // N*128 (reused both layers)
    float* cnt = agg + (size_t)N_NODES * 128;            // N (contiguous after agg)
    float* h   = cnt + N_NODES;                          // N*128

    const int nf1 = 4 * 8 * 256;   // layer1 fragments: 4 ksteps * 8 tiles * 32 lanes * 8
    const int nf2 = 4 * 3 * 256;   // layer2 fragments: 4 ksteps * 3 tiles * 32 lanes * 8
    unsigned int* W1lf = (unsigned int*)(h + (size_t)N_NODES * 128);
    unsigned int* W1rf = W1lf + nf1;
    unsigned int* W2lf = W1rf + nf1;
    unsigned int* W2rf = W2lf + nf2;

    const int nAgg = N_NODES * 128;
    const int scatThreads = N_EDGES * 32;

    // weight pre-pack into fragment order (tiny, once per launch)
    pack_weights_frag<<<(nf1 + 255) / 256, 256, 0, stream>>>(W1l, W1lf, HID_FEAT, 8, nf1);
    pack_weights_frag<<<(nf1 + 255) / 256, 256, 0, stream>>>(W1r, W1rf, HID_FEAT, 8, nf1);
    pack_weights_frag<<<(nf2 + 255) / 256, 256, 0, stream>>>(W2l, W2lf, OUT_FEAT, 3, nf2);
    pack_weights_frag<<<(nf2 + 255) / 256, 256, 0, stream>>>(W2r, W2rf, OUT_FEAT, 3, nf2);

    // ---- layer 1 ----
    zero_f32<<<(nAgg + N_NODES + 255) / 256, 256, 0, stream>>>(agg, nAgg + N_NODES);
    scatter_edges<<<(scatThreads + 255) / 256, 256, 0, stream>>>(x, ei, agg, cnt,
                                                                 N_EDGES, 1);
    sage_layer1<<<N_NODES / 16, 256, 0, stream>>>(x, agg, cnt, (const uint4*)W1lf,
                                                  (const uint4*)W1rf, b1, h);

    // ---- layer 2 (degree counts unchanged; reuse cnt) ----
    zero_f32<<<(nAgg + 255) / 256, 256, 0, stream>>>(agg, nAgg);
    scatter_edges<<<(scatThreads + 255) / 256, 256, 0, stream>>>(h, ei, agg, cnt,
                                                                 N_EDGES, 0);
    sage_layer2<<<N_NODES / 16, 128, 0, stream>>>(h, agg, cnt, (const uint4*)W2lf,
                                                  (const uint4*)W2rf, b2, out);
}
